// HGT_init_62783831933123
// MI455X (gfx1250) — compile-verified, matrix operand order checked by reading, not run on hardware
//
#include <hip/hip_runtime.h>
#include <math.h>

#define HID 64
#define NHEAD 8
#define DHEAD 8

typedef __attribute__((ext_vector_type(16))) __bf16 v16bf;
typedef __attribute__((ext_vector_type(4)))  __bf16 v4bf;
typedef __attribute__((ext_vector_type(8)))  float  v8f;

// ---------------------------------------------------------------------------
// Monotone uint key encoding for float atomic-max
// ---------------------------------------------------------------------------
__device__ __forceinline__ unsigned fkey(float x) {
    unsigned u = __float_as_uint(x);
    return (u & 0x80000000u) ? ~u : (u | 0x80000000u);
}
__device__ __forceinline__ float funkey(unsigned k) {
    unsigned u = (k & 0x80000000u) ? (k & 0x7FFFFFFFu) : ~k;
    return __uint_as_float(u);
}

__device__ __forceinline__ float gelu_exact(float v) {
    return 0.5f * v * (1.0f + erff(v * 0.70710678118654752f));
}

// ---------------------------------------------------------------------------
// Generic WMMA GEMM: C[M,64] = epilogue(A[M,K] @ W[64,K]^T + bias)
// MODE 0: stem   -> relu(.)                 stored to C (ldc)
// MODE 1: proj   -> (.)                     stored to C (ldc)
// MODE 2: update -> A pre-activated w/ exact GELU; out = beta*(.) + (1-beta)*h
//                   stored to h_io (stride 64) AND out2 (stride ldo)
// Block: 256 threads = 8 waves; each wave computes a 16x64 slab.
// ---------------------------------------------------------------------------
template<int MODE>
__global__ __launch_bounds__(256)
void gemm64_wmma_kernel(const float* __restrict__ A, int M, int K,
                        const float* __restrict__ W,      // [64, K]
                        const float* __restrict__ bias,   // [64]
                        float* __restrict__ C, int ldc,
                        float* __restrict__ h_io,         // MODE 2
                        float* __restrict__ out2, int ldo,// MODE 2
                        const float* __restrict__ skip_scalar)
{
    __shared__ __bf16 As[128][32];
    __shared__ __bf16 Ws[64][32];

    const int tid     = threadIdx.x;
    const int wave    = tid >> 5;
    const int lane    = tid & 31;
    const int laneLo  = lane & 15;
    const int hiHalf  = (lane >> 4) << 3;   // 0 for lanes 0-15, 8 for 16-31
    const int rowBase = blockIdx.x * 128;
    const int waveRow = wave * 16;
    const bool fullTile = (rowBase + 128 <= M);

    v8f acc[4] = {};

    for (int k0 = 0; k0 < K; k0 += 32) {
        // ---- stage A tile (128x32) as bf16: 4 x float4 per thread ----
        float4 av[4];
        #pragma unroll
        for (int i = 0; i < 4; ++i) {
            int idx = i * 256 + tid;            // 0..1023
            int r = idx >> 3, q = idx & 7;      // row, float4-within-row
            int gr = rowBase + r;
            if (gr > M - 1) gr = M - 1;         // clamp: branch-free OOB handling
            av[i] = *(const float4*)(A + (size_t)gr * K + k0 + q * 4);
        }
        #pragma unroll
        for (int i = 0; i < 4; ++i) {
            int idx = i * 256 + tid;
            int r = idx >> 3, q = idx & 7;
            float4 v = av[i];
            if (MODE == 2) {
                v.x = gelu_exact(v.x); v.y = gelu_exact(v.y);
                v.z = gelu_exact(v.z); v.w = gelu_exact(v.w);
            }
            v4bf p;
            p[0] = (__bf16)v.x; p[1] = (__bf16)v.y;
            p[2] = (__bf16)v.z; p[3] = (__bf16)v.w;
            *(v4bf*)&As[r][q * 4] = p;
        }
        // ---- stage W tile (64x32) as bf16: 2 x float4 per thread ----
        float4 wv[2];
        #pragma unroll
        for (int i = 0; i < 2; ++i) {
            int idx = i * 256 + tid;            // 0..511
            int r = idx >> 3, q = idx & 7;
            wv[i] = *(const float4*)(W + (size_t)r * K + k0 + q * 4);
        }
        #pragma unroll
        for (int i = 0; i < 2; ++i) {
            int idx = i * 256 + tid;
            int r = idx >> 3, q = idx & 7;
            float4 v = wv[i];
            v4bf p;
            p[0] = (__bf16)v.x; p[1] = (__bf16)v.y;
            p[2] = (__bf16)v.z; p[3] = (__bf16)v.w;
            *(v4bf*)&Ws[r][q * 4] = p;
        }
        __syncthreads();

        // ---- build A fragment per ISA 16-bit 16x32 layout ----
        v16bf aF;
        #pragma unroll
        for (int i = 0; i < 16; ++i) {
            int kk = (i < 8 ? i : i + 8) + hiHalf;
            aF[i] = As[waveRow + laneLo][kk];
        }
        // ---- 4 column tiles of 16 ----
        #pragma unroll
        for (int nt = 0; nt < 4; ++nt) {
            v16bf bF;
            #pragma unroll
            for (int i = 0; i < 16; ++i) {
                int kk = (i < 8 ? i : i + 8) + hiHalf;
                bF[i] = Ws[nt * 16 + laneLo][kk];
            }
            acc[nt] = __builtin_amdgcn_wmma_f32_16x16x32_bf16(
                false, aF, false, bF, (short)0, acc[nt], false, false);
        }
        __syncthreads();
    }

    float beta = 0.0f;
    if (MODE == 2) beta = 1.0f / (1.0f + expf(-skip_scalar[0]));

    #pragma unroll
    for (int nt = 0; nt < 4; ++nt) {
        int n  = nt * 16 + laneLo;
        float bn = bias[n];
        #pragma unroll
        for (int r = 0; r < 8; ++r) {
            int m = rowBase + waveRow + r + hiHalf;   // D layout: M = r (+8 hi lanes)
            if (fullTile || m < M) {
                float v = acc[nt][r] + bn;
                if (MODE == 0) v = fmaxf(v, 0.0f);
                if (MODE == 2) {
                    float hp = h_io[(size_t)m * HID + n];
                    v = beta * v + (1.0f - beta) * hp;
                    h_io[(size_t)m * HID + n] = v;
                    out2[(size_t)m * ldo + n] = v;
                } else {
                    C[(size_t)m * ldc + n] = v;
                }
            }
        }
    }
}

// ---------------------------------------------------------------------------
// Fuse per-head relation projection into the projection weight:
//   Wout[h*8+e][f] = sum_d W[h*8+d][f] * rel[h][d][e]
//   bout[h*8+e]    = sum_d b[h*8+d]    * rel[h][d][e]
// grid = 64 blocks (output row), block = 64 threads (f)
// ---------------------------------------------------------------------------
__global__ void fuse_weights_kernel(const float* __restrict__ W,   // [64][64]
                                    const float* __restrict__ b,   // [64]
                                    const float* __restrict__ rel, // [8][8][8]
                                    float* __restrict__ Wout,
                                    float* __restrict__ bout)
{
    int o = blockIdx.x;           // h*8 + e
    int f = threadIdx.x;          // 0..63
    int h = o >> 3, e = o & 7;
    float acc = 0.0f;
    #pragma unroll
    for (int d = 0; d < DHEAD; ++d)
        acc += W[(size_t)(h * 8 + d) * HID + f] * rel[h * 64 + d * 8 + e];
    Wout[(size_t)o * HID + f] = acc;
    if (f == 0) {
        float bb = 0.0f;
        #pragma unroll
        for (int d = 0; d < DHEAD; ++d)
            bb += b[h * 8 + d] * rel[h * 64 + d * 8 + e];
        bout[o] = bb;
    }
}

// ---------------------------------------------------------------------------
// Edge pass 1: logits + segment max (thread per (edge, head))
// ---------------------------------------------------------------------------
__global__ __launch_bounds__(256)
void edge_logit_kernel(const int* __restrict__ src, const int* __restrict__ dst,
                       const float* __restrict__ q,   // [Ndst,64] dst-side
                       const float* __restrict__ kr,  // [Nsrc,64] src-side
                       const float* __restrict__ prel,// [8]
                       float* __restrict__ logits,    // [E,8]
                       unsigned* __restrict__ mkey,   // [Ndst,8]
                       int E)
{
    int gid = blockIdx.x * blockDim.x + threadIdx.x;
    if (gid >= E * NHEAD) return;
    int e = gid >> 3, h = gid & 7;
    __builtin_prefetch(&src[e + 2048], 0, 0);
    __builtin_prefetch(&dst[e + 2048], 0, 0);
    int s = src[e], d = dst[e];
    const float4* qp = (const float4*)(q  + (size_t)d * HID + h * DHEAD);
    const float4* kp = (const float4*)(kr + (size_t)s * HID + h * DHEAD);
    float4 qa = qp[0], qb = qp[1];
    float4 ka = kp[0], kb_ = kp[1];
    float acc = qa.x * ka.x + qa.y * ka.y + qa.z * ka.z + qa.w * ka.w
              + qb.x * kb_.x + qb.y * kb_.y + qb.z * kb_.z + qb.w * kb_.w;
    float logit = acc * prel[h] * 0.35355339059327373f;  // 1/sqrt(DH)
    logits[gid] = logit;
    atomicMax(&mkey[(size_t)d * NHEAD + h], fkey(logit));
}

// ---------------------------------------------------------------------------
// Edge pass 2: w = exp(logit - max); accumulate denom and unnormalized message
// ---------------------------------------------------------------------------
__global__ __launch_bounds__(256)
void edge_accum_kernel(const int* __restrict__ src, const int* __restrict__ dst,
                       const float* __restrict__ logits,
                       const unsigned* __restrict__ mkey,
                       const float* __restrict__ vr,   // [Nsrc,64]
                       float* __restrict__ ssum,       // [Ndst,8]
                       float* __restrict__ o,          // [Ndst,64]
                       int E)
{
    int gid = blockIdx.x * blockDim.x + threadIdx.x;
    if (gid >= E * NHEAD) return;
    int e = gid >> 3, h = gid & 7;
    __builtin_prefetch(&src[e + 2048], 0, 0);
    int s = src[e], d = dst[e];
    float m = funkey(mkey[(size_t)d * NHEAD + h]);
    float w = expf(logits[gid] - m);
    atomicAdd(&ssum[(size_t)d * NHEAD + h], w);
    const float4* vp = (const float4*)(vr + (size_t)s * HID + h * DHEAD);
    float4 va = vp[0], vb = vp[1];
    float* op = o + (size_t)d * HID + h * DHEAD;
    atomicAdd(&op[0], w * va.x); atomicAdd(&op[1], w * va.y);
    atomicAdd(&op[2], w * va.z); atomicAdd(&op[3], w * va.w);
    atomicAdd(&op[4], w * vb.x); atomicAdd(&op[5], w * vb.y);
    atomicAdd(&op[6], w * vb.z); atomicAdd(&op[7], w * vb.w);
}

// ---------------------------------------------------------------------------
// Node pass: normalize message by softmax denominator
// ---------------------------------------------------------------------------
__global__ __launch_bounds__(256)
void finalize_kernel(float* __restrict__ o, const float* __restrict__ ssum, int N)
{
    int gid = blockIdx.x * blockDim.x + threadIdx.x;
    if (gid >= N * HID) return;
    int n = gid >> 6;
    int h = (gid >> 3) & 7;
    float s = ssum[(size_t)n * NHEAD + h];
    o[gid] = (s > 0.0f) ? (o[gid] / s) : 0.0f;
}

// ---------------------------------------------------------------------------
extern "C" void kernel_launch(void* const* d_in, const int* in_sizes, int n_in,
                              void* d_out, int out_size, void* d_ws, size_t ws_size,
                              hipStream_t stream)
{
    const float* x1     = (const float*)d_in[0];
    const float* x2     = (const float*)d_in[1];
    const int*   edge12 = (const int*)d_in[2];
    const int*   edge21 = (const int*)d_in[3];
    const float* w_in   = (const float*)d_in[4];
    const float* b_in   = (const float*)d_in[5];
    const float* kw     = (const float*)d_in[6];
    const float* kb     = (const float*)d_in[7];
    const float* qw     = (const float*)d_in[8];
    const float* qb     = (const float*)d_in[9];
    const float* vw     = (const float*)d_in[10];
    const float* vb     = (const float*)d_in[11];
    const float* aw     = (const float*)d_in[12];
    const float* ab     = (const float*)d_in[13];
    const float* skip   = (const float*)d_in[14];
    const float* a_rel  = (const float*)d_in[15];
    const float* m_rel  = (const float*)d_in[16];
    const float* p_rel  = (const float*)d_in[17];

    const int F  = 256;
    const int N1 = in_sizes[0] / F;
    const int N2 = in_sizes[1] / F;
    const int E  = in_sizes[2] / 2;
    const int L  = 2;
    const int OUTW = HID * L;   // 128

    // ---- workspace carve-up ----
    float* ws = (float*)d_ws;
    size_t off = 0;
    auto alloc = [&](size_t n) { float* p = ws + off; off += n; return p; };
    float* h1  = alloc((size_t)N1 * HID);
    float* h2  = alloc((size_t)N2 * HID);
    float* q1  = alloc((size_t)N1 * HID);
    float* kr1 = alloc((size_t)N1 * HID);
    float* vr1 = alloc((size_t)N1 * HID);
    float* q2  = alloc((size_t)N2 * HID);
    float* kr2 = alloc((size_t)N2 * HID);
    float* vr2 = alloc((size_t)N2 * HID);
    float* o1  = alloc((size_t)N1 * HID);
    float* o2  = alloc((size_t)N2 * HID);
    float* s1  = alloc((size_t)N1 * NHEAD);
    float* s2  = alloc((size_t)N2 * NHEAD);
    unsigned* mk1 = (unsigned*)alloc((size_t)N1 * NHEAD);
    unsigned* mk2 = (unsigned*)alloc((size_t)N2 * NHEAD);
    float* lg  = alloc((size_t)E * NHEAD);
    float* wK  = alloc(64 * 64);
    float* bK  = alloc(64);
    float* wV  = alloc(64 * 64);
    float* bV  = alloc(64);

    float* Em = (float*)d_out;                         // [N1, 128]
    float* Ed = (float*)d_out + (size_t)N1 * OUTW;     // [N2, 128]

    dim3 blk(256);
    const int g1 = (N1 + 127) / 128;
    const int g2 = (N2 + 127) / 128;
    const int eg = (E * NHEAD + 255) / 256;

    // ---- input stem: h = relu(x @ w_in^T + b_in) ----
    gemm64_wmma_kernel<0><<<g1, blk, 0, stream>>>(x1, N1, F, w_in,           b_in,      h1, HID, nullptr, nullptr, 0, nullptr);
    gemm64_wmma_kernel<0><<<g2, blk, 0, stream>>>(x2, N2, F, w_in + 64 * F,  b_in + 64, h2, HID, nullptr, nullptr, 0, nullptr);

    for (int l = 0; l < L; ++l) {
        const size_t w0 = ((size_t)l * 2 + 0) * 64 * 64, b0 = ((size_t)l * 2 + 0) * 64;
        const size_t w1 = ((size_t)l * 2 + 1) * 64 * 64, b1 = ((size_t)l * 2 + 1) * 64;
        const size_t r0 = ((size_t)l * 2 + 0) * 512,     r1 = ((size_t)l * 2 + 1) * 512;

        // ===== direction 1 -> 2 : src type0 (k1,v1 w/ rel[l,0]), dst type1 (q2) =====
        fuse_weights_kernel<<<64, 64, 0, stream>>>(kw + w0, kb + b0, a_rel + r0, wK, bK);
        fuse_weights_kernel<<<64, 64, 0, stream>>>(vw + w0, vb + b0, m_rel + r0, wV, bV);
        gemm64_wmma_kernel<1><<<g1, blk, 0, stream>>>(h1, N1, HID, wK,       bK,      kr1, HID, nullptr, nullptr, 0, nullptr);
        gemm64_wmma_kernel<1><<<g1, blk, 0, stream>>>(h1, N1, HID, wV,       bV,      vr1, HID, nullptr, nullptr, 0, nullptr);
        gemm64_wmma_kernel<1><<<g2, blk, 0, stream>>>(h2, N2, HID, qw + w1,  qb + b1, q2,  HID, nullptr, nullptr, 0, nullptr);
        hipMemsetAsync(o2,  0, (size_t)N2 * HID   * sizeof(float), stream);
        hipMemsetAsync(s2,  0, (size_t)N2 * NHEAD * sizeof(float), stream);
        hipMemsetAsync(mk2, 0, (size_t)N2 * NHEAD * sizeof(unsigned), stream);
        edge_logit_kernel<<<eg, blk, 0, stream>>>(edge12, edge12 + E, q2, kr1, p_rel + (l * 2 + 0) * 8, lg, mk2, E);
        edge_accum_kernel<<<eg, blk, 0, stream>>>(edge12, edge12 + E, lg, mk2, vr1, s2, o2, E);
        finalize_kernel<<<(N2 * HID + 255) / 256, blk, 0, stream>>>(o2, s2, N2);

        // ===== direction 2 -> 1 : src type1 (k2,v2 w/ rel[l,1]), dst type0 (q1) =====
        fuse_weights_kernel<<<64, 64, 0, stream>>>(kw + w1, kb + b1, a_rel + r1, wK, bK);
        fuse_weights_kernel<<<64, 64, 0, stream>>>(vw + w1, vb + b1, m_rel + r1, wV, bV);
        gemm64_wmma_kernel<1><<<g2, blk, 0, stream>>>(h2, N2, HID, wK,       bK,      kr2, HID, nullptr, nullptr, 0, nullptr);
        gemm64_wmma_kernel<1><<<g2, blk, 0, stream>>>(h2, N2, HID, wV,       bV,      vr2, HID, nullptr, nullptr, 0, nullptr);
        gemm64_wmma_kernel<1><<<g1, blk, 0, stream>>>(h1, N1, HID, qw + w0,  qb + b0, q1,  HID, nullptr, nullptr, 0, nullptr);
        hipMemsetAsync(o1,  0, (size_t)N1 * HID   * sizeof(float), stream);
        hipMemsetAsync(s1,  0, (size_t)N1 * NHEAD * sizeof(float), stream);
        hipMemsetAsync(mk1, 0, (size_t)N1 * NHEAD * sizeof(unsigned), stream);
        edge_logit_kernel<<<eg, blk, 0, stream>>>(edge21, edge21 + E, q1, kr2, p_rel + (l * 2 + 1) * 8, lg, mk1, E);
        edge_accum_kernel<<<eg, blk, 0, stream>>>(edge21, edge21 + E, lg, mk1, vr2, s1, o1, E);
        finalize_kernel<<<(N1 * HID + 255) / 256, blk, 0, stream>>>(o1, s1, N1);

        // ===== updates (after BOTH messages): h = beta*(gelu(o)@aw^T+ab) + (1-beta)*h =====
        gemm64_wmma_kernel<2><<<g1, blk, 0, stream>>>(o1, N1, HID, aw + w0, ab + b0, nullptr, 0, h1, Em + (size_t)l * HID, OUTW, skip + l * 2 + 0);
        gemm64_wmma_kernel<2><<<g2, blk, 0, stream>>>(o2, N2, HID, aw + w1, ab + b1, nullptr, 0, h2, Ed + (size_t)l * HID, OUTW, skip + l * 2 + 1);
    }
}
